// SimpleCNN_87196426043827
// MI455X (gfx1250) — matrix operand order repaired and checked by reading, not compiled, not run on hardware
//
#include <hip/hip_runtime.h>

#define BATCH 256

typedef __attribute__((ext_vector_type(16))) _Float16 v16h;
typedef __attribute__((ext_vector_type(8)))  float    v8f;
typedef unsigned int u32x4 __attribute__((ext_vector_type(4)));
typedef int          i32x4 __attribute__((ext_vector_type(4)));
typedef int          i32x8 __attribute__((ext_vector_type(8)));

union FragH16 { v16h v; _Float16 h[16]; uint4 q[2]; };
union Half8   { _Float16 h[8]; uint4 q; };

#if defined(__has_builtin)
#if __has_builtin(__builtin_amdgcn_tensor_load_to_lds)
#define HAVE_TDM 1
#endif
#endif

// ---------------------------------------------------------------------------
// Mask stages (VALU). m = (windowed sum of 0/1 map) > 1.0  <=>  count >= 2
// ---------------------------------------------------------------------------
__global__ void k_mask1(const int* __restrict__ cmap, unsigned char* __restrict__ m1) {
    int idx = blockIdx.x * blockDim.x + threadIdx.x;
    if (idx >= BATCH * 64 * 64) return;
    int x = idx & 63, y = (idx >> 6) & 63, b = idx >> 12;
    int s = 0;
    for (int dy = 0; dy < 3; ++dy) {
        int yy = y + dy; if (yy >= 64) break;
        for (int dx = 0; dx < 3; ++dx) {
            int xx = x + dx; if (xx >= 64) break;
            s += cmap[(b * 64 + yy) * 64 + xx];
        }
    }
    m1[idx] = (unsigned char)(s > 1);
}

// dst[B][hw][hw] = (2x2 sum of src[B][2hw][2hw]) > 1
__global__ void k_pool_mask(const unsigned char* __restrict__ src,
                            unsigned char* __restrict__ dst, int hw) {
    int idx = blockIdx.x * blockDim.x + threadIdx.x;
    if (idx >= BATCH * hw * hw) return;
    int x = idx % hw, y = (idx / hw) % hw, b = idx / (hw * hw);
    int W = 2 * hw;
    const unsigned char* p = src + ((size_t)b * W + 2 * y) * W + 2 * x;
    int s = p[0] + p[1] + p[W] + p[W + 1];
    dst[idx] = (unsigned char)(s > 1);
}

// dst = (3x3 end-clamped window sum of src) > 1   (both [B][hw][hw])
__global__ void k_win3_mask(const unsigned char* __restrict__ src,
                            unsigned char* __restrict__ dst, int hw) {
    int idx = blockIdx.x * blockDim.x + threadIdx.x;
    if (idx >= BATCH * hw * hw) return;
    int x = idx % hw, y = (idx / hw) % hw, b = idx / (hw * hw);
    int s = 0;
    for (int dy = 0; dy < 3; ++dy) {
        int yy = y + dy; if (yy >= hw) break;
        for (int dx = 0; dx < 3; ++dx) {
            int xx = x + dx; if (xx >= hw) break;
            s += src[((size_t)b * hw + yy) * hw + xx];
        }
    }
    dst[idx] = (unsigned char)(s > 1);
}

// ---------------------------------------------------------------------------
// conv1 (1->32ch, pad=1) gated by m1, ReLU, maxpool2 gated by m2.
// Output p1h in NHWC f16: p1h[b][py][px][c], 32 channels contiguous.
// ---------------------------------------------------------------------------
__global__ void k_conv1_pool(const float* __restrict__ x, const float* __restrict__ W1,
                             const float* __restrict__ b1,
                             const unsigned char* __restrict__ m1,
                             const unsigned char* __restrict__ m2,
                             _Float16* __restrict__ p1h) {
    int idx = blockIdx.x * blockDim.x + threadIdx.x;
    if (idx >= BATCH * 32 * 32 * 32) return;
    int c = idx & 31, px = (idx >> 5) & 31, py = (idx >> 10) & 31, b = idx >> 15;
    float out = 0.f;
    if (m2[((size_t)b * 32 + py) * 32 + px]) {
        float w[9];
        for (int t = 0; t < 9; ++t) w[t] = W1[c * 9 + t];
        float bias = b1[c];
        float best = 0.f;
        for (int sy = 0; sy < 2; ++sy)
        for (int sx = 0; sx < 2; ++sx) {
            int y = 2 * py + sy, xx = 2 * px + sx;
            if (m1[((size_t)b * 64 + y) * 64 + xx]) {
                float acc = bias;
                for (int dy = 0; dy < 3; ++dy) {
                    int iy = y + dy - 1; if (iy < 0 || iy >= 64) continue;
                    for (int dx = 0; dx < 3; ++dx) {
                        int ix = xx + dx - 1; if (ix < 0 || ix >= 64) continue;
                        acc += w[dy * 3 + dx] * x[((size_t)b * 64 + iy) * 64 + ix];
                    }
                }
                float r = acc > 0.f ? acc : 0.f;
                best = r > best ? r : best;
            }
        }
        out = best;
    }
    p1h[idx] = (_Float16)out;
}

// ---------------------------------------------------------------------------
// Pack W2[64][32][3][3] f32 -> per-(tap,group,lane) B-fragments f16.
// Lane l: N = l&15 (out ch within group), elements j: K(in ch) = 8*(l>>4)+j+(j&8).
// ---------------------------------------------------------------------------
__global__ void k_pack_w2(const float* __restrict__ W2, _Float16* __restrict__ wf) {
    int idx = blockIdx.x * blockDim.x + threadIdx.x;
    if (idx >= 9 * 4 * 32 * 16) return;
    int j = idx & 15, lane = (idx >> 4) & 31, g = (idx >> 9) & 3, tap = idx >> 11;
    int oc = g * 16 + (lane & 15);
    int ic = 8 * (lane >> 4) + j + (j & 8);
    wf[idx] = (_Float16)W2[(size_t)(oc * 32 + ic) * 9 + tap];
}

// ---------------------------------------------------------------------------
// conv2 via WMMA f32<=f16 16x16x32. One wave = 16 pixels x 16 outch tile.
// 9 taps -> 9 WMMAs, K=32 input channels each. m3-gated ReLU epilogue.
// OOB x handled by clamped address + data select (global_load, no scratch).
// r2h stored NCHW f16: r2h[b][oc][y][x].
// ---------------------------------------------------------------------------
__global__ void __launch_bounds__(32) k_conv2(const _Float16* __restrict__ p1h,
                                              const _Float16* __restrict__ wf,
                                              const float* __restrict__ b2,
                                              const unsigned char* __restrict__ m3,
                                              _Float16* __restrict__ r2h) {
    int l    = threadIdx.x;
    int bid  = blockIdx.x;
    int g    = bid & 3;
    int xseg = (bid >> 2) & 1;
    int y    = (bid >> 3) & 31;
    int b    = bid >> 8;
    int x0   = xseg * 16;
    int mrow = l & 15;
    int hi   = l >> 4;
    const uint4 z4 = {0u, 0u, 0u, 0u};

    v8f acc = {};
    for (int ty = 0; ty < 3; ++ty) {
        int yin = y + ty - 1;
        if (yin < 0 || yin >= 32) continue;          // wave-uniform
        const uint4* rowbase = (const uint4*)(p1h + ((size_t)(b * 32 + yin) * 32) * 32);
        for (int tx = 0; tx < 3; ++tx) {
            int xin = x0 + mrow + tx - 1;            // per-lane (per M row)
            bool inb = (xin >= 0) && (xin < 32);
            int  xc  = xin < 0 ? 0 : (xin > 31 ? 31 : xin);   // always-valid address
            const uint4* src = rowbase + (size_t)xc * 4;      // 32 halfs = 4 uint4
            uint4 t0 = src[hi];                               // channels 8*hi..8*hi+7
            uint4 t1 = src[2 + hi];                           // channels 16+8*hi..
            FragH16 A;
            A.q[0] = inb ? t0 : z4;                           // select DATA, not address
            A.q[1] = inb ? t1 : z4;
            int tap = ty * 3 + tx;
            FragH16 Bf;
            const uint4* wsrc = (const uint4*)(wf + ((size_t)(tap * 4 + g) * 32 + l) * 16);
            Bf.q[0] = wsrc[0];
            Bf.q[1] = wsrc[1];
            acc = __builtin_amdgcn_wmma_f32_16x16x32_f16(false, A.v, false, Bf.v,
                                                         (short)0, acc, false, false);
        }
    }
    // Epilogue: D element r -> (M = r + 8*hi -> x = x0 + 8*hi + r, N = l&15 -> oc)
    int oc = g * 16 + (l & 15);
    float bias = b2[oc];
    const unsigned char* m3row = m3 + ((size_t)b * 32 + y) * 32;
    int xbase = x0 + 8 * hi;
    Half8 out;
    for (int r = 0; r < 8; ++r) {
        float v = acc[r] + bias;
        v = v > 0.f ? v : 0.f;
        out.h[r] = (_Float16)(m3row[xbase + r] ? v : 0.f);
    }
    *(uint4*)(r2h + (((size_t)(b * 64 + oc) * 32 + y) * 32 + xbase)) = out.q;
}

// ---------------------------------------------------------------------------
// maxpool2 gated by m4 -> p2h flat f16 in NCHW flatten order [b][c*256+py*16+px]
// ---------------------------------------------------------------------------
__global__ void k_pool2(const _Float16* __restrict__ r2h,
                        const unsigned char* __restrict__ m4,
                        _Float16* __restrict__ p2h) {
    int idx = blockIdx.x * blockDim.x + threadIdx.x;
    if (idx >= BATCH * 64 * 16 * 16) return;
    int px = idx & 15, py = (idx >> 4) & 15, c = (idx >> 8) & 63, b = idx >> 14;
    float v = 0.f;
    if (m4[((size_t)b * 16 + py) * 16 + px]) {
        const _Float16* p = r2h + (((size_t)(b * 64 + c) * 32 + 2 * py) * 32 + 2 * px);
        float a0 = (float)p[0], a1 = (float)p[1], a2 = (float)p[32], a3 = (float)p[33];
        v = a0; v = a1 > v ? a1 : v; v = a2 > v ? a2 : v; v = a3 > v ? a3 : v;
    }
    p2h[(size_t)b * 16384 + c * 256 + py * 16 + px] = (_Float16)v;
}

// ---------------------------------------------------------------------------
// Pack Wfc1[128][16384] f32 -> per-(ktile,group,lane) B-fragments f16.
// ---------------------------------------------------------------------------
__global__ void k_pack_fc1(const float* __restrict__ Wfc1, _Float16* __restrict__ wf) {
    int idx = blockIdx.x * blockDim.x + threadIdx.x;
    if (idx >= 512 * 8 * 32 * 16) return;
    int j = idx & 15, lane = (idx >> 4) & 31, g = (idx >> 9) & 7, kt = idx >> 12;
    int n = g * 16 + (lane & 15);
    int k = kt * 32 + 8 * (lane >> 4) + j + (j & 8);
    wf[idx] = (_Float16)Wfc1[(size_t)n * 16384 + k];
}

// ---------------------------------------------------------------------------
// fc1 via WMMA: M=batch tile(16), N=hidden group(16), K loop 512 x (K=32).
// Weight fragments are staged through LDS by the Tensor Data Mover:
//   per 16-ktile chunk, one tensor_load_to_lds DMAs a 2D tile
//   (16 rows x 1KB, row stride 8KB picks out this wave's N-group),
//   fenced with s_wait_tensorcnt; inner loop reads B via ds_load_b128.
// h[b][n] = relu(p2 . Wfc1^T + bfc1), f32.
// ---------------------------------------------------------------------------
__global__ void __launch_bounds__(32) k_fc1(const _Float16* __restrict__ p2h,
                                            const _Float16* __restrict__ wf,
                                            const float* __restrict__ bfc1,
                                            float* __restrict__ h) {
    int l  = threadIdx.x;
    int mt = blockIdx.x >> 3;
    int g  = blockIdx.x & 7;
    int hi = l >> 4;
    int row = mt * 16 + (l & 15);
    const _Float16* arow = p2h + (size_t)row * 16384;

    v8f acc = {};
#ifdef HAVE_TDM
    __shared__ uint4 ldsB[16 * 64];            // 16 kt * 32 lanes * 32B = 16 KB
    unsigned ldsAddr = (unsigned)(size_t)(&ldsB[0]);
    for (int chunk = 0; chunk < 32; ++chunk) {
        // Global source: fragment block for (kt = chunk*16 + ktl, group g).
        unsigned long long gaddr =
            (unsigned long long)(size_t)wf +
            ((unsigned long long)(chunk * 16 * 8 + g) * 32ull * 16ull) * 2ull;
        // D# group 0: count=1, lds_addr, 57-bit global_addr, type=2.
        u32x4 g0;
        g0[0] = 1u;
        g0[1] = ldsAddr;
        g0[2] = (unsigned)gaddr;
        g0[3] = ((unsigned)((gaddr >> 32) & 0x01FFFFFFull)) | 0x80000000u;
        // D# group 1: data_size=4B; tensor_dim0=256 dw, tensor_dim1=16 rows;
        // tile 256x16 dw; tensor_dim0_stride=2048 dw (=8KB, skips other groups).
        i32x8 g1;
        g1[0] = 0x00020000;            // data_size=2 (4B), mask=0, flags=0
        g1[1] = (int)(256u << 16);     // tensor_dim0[15:0] in [31:16]
        g1[2] = (int)(16u << 16);      // tensor_dim0 hi=0, tensor_dim1 lo=16
        g1[3] = (int)(256u << 16);     // tensor_dim1 hi=0, tile_dim0=256
        g1[4] = 16;                    // tile_dim1=16, tile_dim2=0
        g1[5] = 2048;                  // tensor_dim0_stride lo
        g1[6] = 0;                     // stride hi, dim1_stride lo
        g1[7] = 0;
        i32x4 z0 = {0, 0, 0, 0};
        // WAR fence: previous chunk's ds reads must land before TDM overwrites LDS
        asm volatile("s_wait_dscnt 0x0" ::: "memory");
#if __clang_major__ >= 23
        i32x8 z8 = {0, 0, 0, 0, 0, 0, 0, 0};
        __builtin_amdgcn_tensor_load_to_lds(g0, g1, z0, z0, z8, 0);
#else
        __builtin_amdgcn_tensor_load_to_lds(g0, g1, z0, z0, 0);
#endif
        __builtin_amdgcn_s_wait_tensorcnt(0);
        const uint4* lbase = (const uint4*)&ldsB[0];
        for (int ktl = 0; ktl < 16; ++ktl) {
            int kt = chunk * 16 + ktl;
            const uint4* a0 = (const uint4*)(arow + kt * 32);
            if (kt + 8 < 512) __builtin_prefetch(arow + (kt + 8) * 32, 0, 1);
            FragH16 A;
            A.q[0] = a0[hi];                   // k = kt*32 + 8*hi .. +7
            A.q[1] = a0[2 + hi];               // k = kt*32 + 16 + 8*hi ..
            FragH16 Bf;
            const uint4* lb = lbase + (ktl * 32 + l) * 2;
            Bf.q[0] = lb[0];
            Bf.q[1] = lb[1];
            acc = __builtin_amdgcn_wmma_f32_16x16x32_f16(false, A.v, false, Bf.v,
                                                         (short)0, acc, false, false);
        }
    }
#else
    for (int kt = 0; kt < 512; ++kt) {
        const uint4* a0 = (const uint4*)(arow + kt * 32);
        FragH16 A;
        A.q[0] = a0[hi];
        A.q[1] = a0[2 + hi];
        FragH16 Bf;
        const uint4* wsrc = (const uint4*)(wf + ((size_t)(kt * 8 + g) * 32 + l) * 16);
        Bf.q[0] = wsrc[0];
        Bf.q[1] = wsrc[1];
        acc = __builtin_amdgcn_wmma_f32_16x16x32_f16(false, A.v, false, Bf.v,
                                                     (short)0, acc, false, false);
    }
#endif
    int n = g * 16 + (l & 15);
    float bias = bfc1[n];
    for (int r = 0; r < 8; ++r) {
        int batch = mt * 16 + r + 8 * hi;
        float v = acc[r] + bias;
        h[(size_t)batch * 128 + n] = v > 0.f ? v : 0.f;
    }
}

// ---------------------------------------------------------------------------
// fc2 (tiny): out[b][o] = h[b] . Wfc2[o] + bfc2[o]
// ---------------------------------------------------------------------------
__global__ void k_fc2(const float* __restrict__ h, const float* __restrict__ Wfc2,
                      const float* __restrict__ bfc2, float* __restrict__ out) {
    int idx = blockIdx.x * blockDim.x + threadIdx.x;
    if (idx >= BATCH * 10) return;
    int o = idx % 10, b = idx / 10;
    float acc = bfc2[o];
    const float* hr = h + (size_t)b * 128;
    const float* wr = Wfc2 + (size_t)o * 128;
    for (int j = 0; j < 128; ++j) acc += hr[j] * wr[j];
    out[idx] = acc;
}

// ---------------------------------------------------------------------------
extern "C" void kernel_launch(void* const* d_in, const int* in_sizes, int n_in,
                              void* d_out, int out_size, void* d_ws, size_t ws_size,
                              hipStream_t stream) {
    const float* x    = (const float*)d_in[0];
    const int*   cmap = (const int*)d_in[1];
    const float* W1   = (const float*)d_in[2];
    const float* b1   = (const float*)d_in[3];
    const float* W2   = (const float*)d_in[4];
    const float* b2   = (const float*)d_in[5];
    const float* Wfc1 = (const float*)d_in[6];
    const float* bfc1 = (const float*)d_in[7];
    const float* Wfc2 = (const float*)d_in[8];
    const float* bfc2 = (const float*)d_in[9];
    float* out = (float*)d_out;

    // Workspace carve (256B aligned, all sizes already multiples of 256)
    char* w = (char*)d_ws;
    unsigned char* m1 = (unsigned char*)w;  w += (size_t)BATCH * 64 * 64;          // 1 MB
    unsigned char* m2 = (unsigned char*)w;  w += (size_t)BATCH * 32 * 32;          // 256 KB
    unsigned char* m3 = (unsigned char*)w;  w += (size_t)BATCH * 32 * 32;          // 256 KB
    unsigned char* m4 = (unsigned char*)w;  w += (size_t)BATCH * 16 * 16;          // 64 KB
    _Float16* p1h = (_Float16*)w;           w += (size_t)BATCH * 32 * 32 * 32 * 2; // 16 MB
    _Float16* wf2 = (_Float16*)w;           w += (size_t)9 * 4 * 32 * 16 * 2;      // 36 KB
    _Float16* r2h = (_Float16*)w;           w += (size_t)BATCH * 64 * 32 * 32 * 2; // 32 MB
    _Float16* p2h = (_Float16*)w;           w += (size_t)BATCH * 16384 * 2;        // 8 MB
    _Float16* wfc = (_Float16*)w;           w += (size_t)512 * 8 * 32 * 16 * 2;    // 4 MB
    float*    h   = (float*)w;              w += (size_t)BATCH * 128 * 4;          // 128 KB

    const int T = 256;
    // masks
    k_mask1<<<(BATCH * 64 * 64 + T - 1) / T, T, 0, stream>>>(cmap, m1);
    k_pool_mask<<<(BATCH * 32 * 32 + T - 1) / T, T, 0, stream>>>(m1, m2, 32);
    k_win3_mask<<<(BATCH * 32 * 32 + T - 1) / T, T, 0, stream>>>(m2, m3, 32);
    k_pool_mask<<<(BATCH * 16 * 16 + T - 1) / T, T, 0, stream>>>(m3, m4, 16);
    // conv1 + pool1 fused (VALU)
    k_conv1_pool<<<(BATCH * 32 * 32 * 32 + T - 1) / T, T, 0, stream>>>(x, W1, b1, m1, m2, p1h);
    // conv2 on WMMA
    k_pack_w2<<<(9 * 4 * 32 * 16 + T - 1) / T, T, 0, stream>>>(W2, wf2);
    k_conv2<<<BATCH * 32 * 2 * 4, 32, 0, stream>>>(p1h, wf2, b2, m3, r2h);
    // pool2
    k_pool2<<<(BATCH * 64 * 16 * 16 + T - 1) / T, T, 0, stream>>>(r2h, m4, p2h);
    // fc1 on WMMA (+ TDM weight staging)
    k_pack_fc1<<<(512 * 8 * 32 * 16 + T - 1) / T, T, 0, stream>>>(Wfc1, wfc);
    k_fc1<<<16 * 8, 32, 0, stream>>>(p2h, wfc, bfc1, h);
    // fc2
    k_fc2<<<(BATCH * 10 + T - 1) / T, T, 0, stream>>>(h, Wfc2, bfc2, out);

    (void)in_sizes; (void)n_in; (void)out_size; (void)ws_size;
}
